// BandSimVQ_48378511622624
// MI455X (gfx1250) — matrix operand order repaired and checked by reading, not compiled, not run on hardware
//
#include <hip/hip_runtime.h>
#include <stdint.h>

#define NB    8      // batch
#define BANDS 6
#define DD    512    // feature dim
#define TT    768    // frames
#define CS    2048   // codebook size
#define CD    512    // codebook dim

typedef __attribute__((ext_vector_type(16))) __bf16 v16bf;
typedef __attribute__((ext_vector_type(8)))  float  v8f;

union BF16x16 { v16bf v; uint4 q[2]; unsigned short u[16]; };

__device__ __forceinline__ unsigned short f2bf(float f) {
    unsigned int u = __float_as_uint(f);
    u += 0x7FFFu + ((u >> 16) & 1u);          // round-to-nearest-even
    return (unsigned short)(u >> 16);
}

// ---------------------------------------------------------------------------
// K0: bulk convert float -> bf16 (memory bound; keeps conversions out of GEMMs)
// ---------------------------------------------------------------------------
__global__ void convert_bf16(const float* __restrict__ in,
                             unsigned short* __restrict__ out, int n4) {
    int i = blockIdx.x * blockDim.x + threadIdx.x;
    int stride = gridDim.x * blockDim.x;
    for (; i < n4; i += stride) {
        float4 v = ((const float4*)in)[i];
        ushort4 o;
        o.x = f2bf(v.x); o.y = f2bf(v.y); o.z = f2bf(v.z); o.w = f2bf(v.w);
        ((ushort4*)out)[i] = o;
    }
}

// ---------------------------------------------------------------------------
// K1/K2: tiled transpose float -> bf16.  in[z][r][c] -> out[z][c][r]
// ---------------------------------------------------------------------------
__global__ void transpose_to_bf16(const float* __restrict__ in,
                                  unsigned short* __restrict__ out,
                                  int R, int C) {
    __shared__ float tile[32][33];
    const int z  = blockIdx.z;
    const float* ip = in + (size_t)z * R * C;
    unsigned short* op = out + (size_t)z * R * C;
    const int c0 = blockIdx.x * 32, r0 = blockIdx.y * 32;
#pragma unroll
    for (int k = 0; k < 4; k++) {
        int r = r0 + threadIdx.y + k * 8;
        int c = c0 + threadIdx.x;
        tile[threadIdx.y + k * 8][threadIdx.x] = ip[(size_t)r * C + c];
    }
    __syncthreads();
#pragma unroll
    for (int k = 0; k < 4; k++) {
        int c = c0 + threadIdx.y + k * 8;
        int r = r0 + threadIdx.x;
        op[(size_t)c * R + r] = f2bf(tile[threadIdx.x][threadIdx.y + k * 8]);
    }
}

// ---------------------------------------------------------------------------
// K3: implicit[band] = codebooks[band] (CSxCD) @ W[band] (CDxDD), bf16 WMMA.
// Pure loads + WMMA in the inner loop (operands pre-converted to bf16).
// grid (DD/16, CS/128, BANDS), block 256 (8 waves).
// ---------------------------------------------------------------------------
__global__ void gemm_implicit(const unsigned short* __restrict__ cbbf,  // [band][c][k]
                              const unsigned short* __restrict__ wt,    // [band][n][k]
                              float* __restrict__ impf,
                              unsigned short* __restrict__ impbf) {
    const int band = blockIdx.z;
    const int tid  = threadIdx.x;
    const int w    = tid >> 5, lane = tid & 31;
    const int col  = lane & 15, hi = lane >> 4;
    const int c0   = blockIdx.y * 128 + w * 16;
    const int d0   = blockIdx.x * 16;

    const unsigned short* Arow = cbbf + ((size_t)band * CS + c0 + col) * CD;
    const unsigned short* Bcol = wt   + ((size_t)band * DD + d0 + col) * CD;

    v8f acc = {};
#pragma unroll
    for (int kk = 0; kk < CD / 32; kk++) {
        BF16x16 a, b;
        const uint4* ap = (const uint4*)(Arow + kk * 32 + hi * 8);
        a.q[0] = ap[0]; a.q[1] = ap[2];                    // K runs {0..7},{16..23} (+8 hi)
        const uint4* bp = (const uint4*)(Bcol + kk * 32 + hi * 8);
        b.q[0] = bp[0]; b.q[1] = bp[2];                    // K runs {0..7},{16..23} (+8 hi)
        acc = __builtin_amdgcn_wmma_f32_16x16x32_bf16(false, a.v, false, b.v,
                                                      (short)0, acc, false, false);
    }
    const int dcol = d0 + col;
#pragma unroll
    for (int i = 0; i < 8; i++) {
        const int c = c0 + hi * 8 + i;
        const size_t o = ((size_t)band * CS + c) * DD + dcol;
        impf[o]  = acc[i];
        impbf[o] = f2bf(acc[i]);
    }
}

// ---------------------------------------------------------------------------
// K4: c2[row] = sum_d implicit[row][d]^2.  One wave per row.
// ---------------------------------------------------------------------------
__global__ void row_norm(const float* __restrict__ impf, float* __restrict__ c2) {
    const int w = threadIdx.x >> 5, lane = threadIdx.x & 31;
    const int row = blockIdx.x * 8 + w;
    const float* p = impf + (size_t)row * DD;
    float s = 0.f;
#pragma unroll 4
    for (int j = lane; j < DD; j += 32) { float v = p[j]; s += v * v; }
#pragma unroll
    for (int m = 16; m; m >>= 1) s += __shfl_xor(s, m, 32);
    if (lane == 0) c2[row] = s;
}

// ---------------------------------------------------------------------------
// K5: distance + argmin.  score = c2[c] - 2 * <x_t, implicit_c>.
// grid (TT/128, NB*BANDS), block 256 (8 waves x 16 t-rows each).
// A fragments (16 x v16bf = 128 VGPRs/lane) pinned in registers.
// B chunks double-buffered in LDS with software-pipelined staging:
// global loads for chunk cc+1 are issued before the WMMA loop on chunk cc.
// ---------------------------------------------------------------------------
#define BPAD 520
#define NCHUNK (CS / 16)
__global__ void dist_argmin(const unsigned short* __restrict__ xt,     // [bb][t][d]
                            const unsigned short* __restrict__ impbf,  // [band][c][d]
                            const float* __restrict__ c2,
                            int* __restrict__ idxOut,
                            float* __restrict__ idxF) {
    extern __shared__ __align__(16) char smem[];
    unsigned short* Bs0 = (unsigned short*)smem;            // 16 * BPAD bf16
    unsigned short* Bs1 = Bs0 + 16 * BPAD;

    const int bb = blockIdx.y, band = bb % BANDS;
    const int tid = threadIdx.x, w = tid >> 5, lane = tid & 31;
    const int col = lane & 15, hi = lane >> 4;
    const int tw  = blockIdx.x * 128 + w * 16;

    // Load this wave's A fragments once into registers (reused 128x).
    BF16x16 afrag[CD / 32];
    {
        const unsigned short* Arow = xt + ((size_t)bb * TT + tw + col) * DD;
#pragma unroll
        for (int kk = 0; kk < CD / 32; kk++) {
            const uint4* ap = (const uint4*)(Arow + kk * 32 + hi * 8);
            afrag[kk].q[0] = ap[0];
            afrag[kk].q[1] = ap[2];                         // +16 bf16 = +2 uint4
        }
    }

    // chunk cc = 16 codebook rows = 1024 uint4, 4 per thread
    const uint4* bsrcBase = (const uint4*)(impbf + (size_t)band * CS * DD);

    float bestv[8]; int bestc[8];
#pragma unroll
    for (int i = 0; i < 8; i++) { bestv[i] = 3.4e38f; bestc[i] = 0; }

    // prologue: stage chunk 0
    {
        const uint4* s = bsrcBase;
        uint4 r[4];
#pragma unroll
        for (int k = 0; k < 4; k++) r[k] = s[tid + k * 256];
#pragma unroll
        for (int k = 0; k < 4; k++) {
            const int j = tid + k * 256;
            *((uint4*)(Bs0 + (j >> 6) * BPAD) + (j & 63)) = r[k];
        }
    }
    __syncthreads();

    for (int cc = 0; cc < NCHUNK; cc++) {
        unsigned short* cur = (cc & 1) ? Bs1 : Bs0;
        unsigned short* nxt = (cc & 1) ? Bs0 : Bs1;
        const bool more = (cc + 1 < NCHUNK);

        // issue global loads for next chunk before compute (overlaps WMMAs)
        uint4 r[4];
        if (more) {
            const uint4* s = bsrcBase + (size_t)(cc + 1) * 1024;
#pragma unroll
            for (int k = 0; k < 4; k++) r[k] = s[tid + k * 256];
        }

        v8f acc = {};
#pragma unroll
        for (int kk = 0; kk < CD / 32; kk++) {
            BF16x16 b;
            const uint4* bp = (const uint4*)(cur + col * BPAD + kk * 32 + hi * 8);
            b.q[0] = bp[0]; b.q[1] = bp[2];                 // K runs {0..7},{16..23} (+8 hi)
            acc = __builtin_amdgcn_wmma_f32_16x16x32_bf16(false, afrag[kk].v, false, b.v,
                                                          (short)0, acc, false, false);
        }
        const float cn  = c2[(size_t)band * CS + cc * 16 + col];
        const int cidx  = cc * 16 + col;
#pragma unroll
        for (int i = 0; i < 8; i++) {
            const float s = cn - 2.0f * acc[i];
            if (s < bestv[i]) { bestv[i] = s; bestc[i] = cidx; }  // cols ascend per lane
        }

        if (more) {
#pragma unroll
            for (int k = 0; k < 4; k++) {
                const int j = tid + k * 256;
                *((uint4*)(nxt + (j >> 6) * BPAD) + (j & 63)) = r[k];
            }
        }
        __syncthreads();
    }

    // cross-lane argmin over the 16 columns (within each 16-lane half)
#pragma unroll
    for (int m = 1; m <= 8; m <<= 1) {
#pragma unroll
        for (int i = 0; i < 8; i++) {
            const float ov = __shfl_xor(bestv[i], m, 32);
            const int   oc = __shfl_xor(bestc[i], m, 32);
            if (ov < bestv[i] || (ov == bestv[i] && oc < bestc[i])) {
                bestv[i] = ov; bestc[i] = oc;
            }
        }
    }
    if (col == 0) {
#pragma unroll
        for (int i = 0; i < 8; i++) {
            const int t = tw + hi * 8 + i;
            idxOut[(size_t)bb * TT + t] = bestc[i];
            idxF[(size_t)bb * TT + t]   = (float)bestc[i];   // exact: idx < 2048
        }
    }
}

// ---------------------------------------------------------------------------
// K6: gather implicit rows, write quantized [bb][d][t] coalesced along t,
// accumulate deterministic per-block loss partials.  grid (TT/32, NB*BANDS).
// ---------------------------------------------------------------------------
#define QPAD 516
__global__ void gather_out_loss(const float* __restrict__ x,
                                const float* __restrict__ impf,
                                const int* __restrict__ idx,
                                float* __restrict__ outq,
                                float* __restrict__ partial) {
    extern __shared__ __align__(16) char smem[];
    float* Qs  = (float*)smem;                           // 32 * QPAD
    float* red = (float*)(smem + (size_t)32 * QPAD * 4); // 256
    __shared__ int sidx[32];

    const int bb = blockIdx.y, band = bb % BANDS;
    const int t0 = blockIdx.x * 32, tid = threadIdx.x;
    if (tid < 32) sidx[tid] = idx[(size_t)bb * TT + t0 + tid];
    __syncthreads();

    for (int j = tid; j < 32 * 128; j += 256) {
        const int r = j >> 7, c4 = j & 127;
        const uint4* q = (const uint4*)(impf + ((size_t)band * CS + sidx[r]) * DD);
        *((uint4*)(Qs + r * QPAD) + c4) = q[c4];
    }
    __syncthreads();

    float lsum = 0.f;
    const int tt = tid & 31;
    for (int d = tid >> 5; d < DD; d += 8) {
        const float qv = Qs[tt * QPAD + d];
        const size_t go = ((size_t)bb * DD + d) * TT + t0 + tt;
        const float dfv = x[go] - qv;
        lsum += dfv * dfv;
        outq[go] = qv;                 // straight-through output == q numerically
    }
    red[tid] = lsum;
    __syncthreads();
    for (int s = 128; s > 0; s >>= 1) {
        if (tid < s) red[tid] += red[tid + s];
        __syncthreads();
    }
    if (tid == 0) partial[(size_t)blockIdx.y * gridDim.x + blockIdx.x] = red[0];
}

// ---------------------------------------------------------------------------
// K7: final deterministic loss reduction.
// ---------------------------------------------------------------------------
__global__ void final_loss(const float* __restrict__ partial, int n,
                           float* __restrict__ lossOut) {
    __shared__ float red[256];
    float s = 0.f;
    for (int j = threadIdx.x; j < n; j += 256) s += partial[j];
    red[threadIdx.x] = s;
    __syncthreads();
    for (int st = 128; st > 0; st >>= 1) {
        if (threadIdx.x < st) red[threadIdx.x] += red[threadIdx.x + st];
        __syncthreads();
    }
    if (threadIdx.x == 0)
        *lossOut = 1.25f * red[0] / (float)(NB * BANDS * TT);  // commit + 0.25*codebook
}

// ---------------------------------------------------------------------------
extern "C" void kernel_launch(void* const* d_in, const int* in_sizes, int n_in,
                              void* d_out, int out_size, void* d_ws, size_t ws_size,
                              hipStream_t stream) {
    (void)in_sizes; (void)n_in; (void)out_size; (void)ws_size;
    const float* x  = (const float*)d_in[0];   // [8,6,512,768]
    const float* cb = (const float*)d_in[1];   // [6,2048,512]
    const float* W  = (const float*)d_in[2];   // [6,512,512]

    // workspace layout
    char* ws = (char*)d_ws;
    size_t off = 0;
    auto take = [&](size_t bytes) { char* p = ws + off; off = (off + bytes + 255) & ~(size_t)255; return p; };
    unsigned short* xt    = (unsigned short*)take((size_t)NB * BANDS * TT * DD * 2);
    unsigned short* cbbf  = (unsigned short*)take((size_t)BANDS * CS * CD * 2);
    unsigned short* wt    = (unsigned short*)take((size_t)BANDS * DD * CD * 2);
    float*          impf  = (float*)take((size_t)BANDS * CS * DD * 4);
    unsigned short* impbf = (unsigned short*)take((size_t)BANDS * CS * DD * 2);
    float*          c2    = (float*)take((size_t)BANDS * CS * 4);
    int*            idxI  = (int*)take((size_t)NB * BANDS * TT * 4);
    float*          part  = (float*)take((size_t)(NB * BANDS) * (TT / 32) * 4);

    // output layout: quantized | indices(as float) | loss
    float* outq  = (float*)d_out;
    float* idxF  = outq + (size_t)NB * BANDS * DD * TT;
    float* lossP = idxF + (size_t)NB * BANDS * TT;

    // K0: codebooks -> bf16 (row-major already matches A-fragment layout)
    const int cbN4 = (BANDS * CS * CD) / 4;
    convert_bf16<<<dim3(cbN4 / (256 * 4)), 256, 0, stream>>>(cb, cbbf, cbN4);
    // K1: x [bb][d][t] -> xt bf16 [bb][t][d]
    transpose_to_bf16<<<dim3(TT / 32, DD / 32, NB * BANDS), dim3(32, 8), 0, stream>>>(x, xt, DD, TT);
    // K2: W [band][k][n] -> wt bf16 [band][n][k]
    transpose_to_bf16<<<dim3(DD / 32, CD / 32, BANDS), dim3(32, 8), 0, stream>>>(W, wt, CD, DD);
    // K3: implicit codebook GEMM (bf16 WMMA, f32 accumulate)
    gemm_implicit<<<dim3(DD / 16, CS / 128, BANDS), 256, 0, stream>>>(cbbf, wt, impf, impbf);
    // K4: row norms
    row_norm<<<(BANDS * CS) / 8, 256, 0, stream>>>(impf, c2);
    // K5: distance GEMM + argmin (double-buffered B chunks, ~33 KB LDS)
    const size_t smem5 = (size_t)2 * 16 * BPAD * 2;
    dist_argmin<<<dim3(TT / 128, NB * BANDS), 256, smem5, stream>>>(xt, impbf, c2, idxI, idxF);
    // K6: gather + quantized output + loss partials
    const size_t smem6 = (size_t)32 * QPAD * 4 + 256 * 4;
    gather_out_loss<<<dim3(TT / 32, NB * BANDS), 256, smem6, stream>>>(x, impf, idxI, outq, part);
    // K7: final loss
    final_loss<<<1, 256, 0, stream>>>(part, (NB * BANDS) * (TT / 32), lossP);
}
